// MultiHeadSelfAttention_67087389163715
// MI455X (gfx1250) — matrix operand order repaired and checked by reading, not compiled, not run on hardware
//
#include <hip/hip_runtime.h>
#include <hip/hip_bf16.h>

typedef __attribute__((ext_vector_type(16))) _Float16 v16h;
typedef __attribute__((ext_vector_type(8)))  float    v8f;
typedef unsigned int u32x4 __attribute__((ext_vector_type(4)));
typedef int          i32x8 __attribute__((ext_vector_type(8)));
typedef int          i32x4 __attribute__((ext_vector_type(4)));

#define Bx   2
#define Sx   2048
#define Dx   2048
#define Hx   16
#define DKx  128
#define Mtot (Bx * Sx)   // 4096

#if __has_builtin(__builtin_amdgcn_tensor_load_to_lds)
#define HAVE_TDM 1
#else
#define HAVE_TDM 0
#endif

// ---------------------------------------------------------------------------
// CDNA5 async global->LDS copy (ASYNCcnt path) + wait  (fallback path)
// ---------------------------------------------------------------------------
__device__ __forceinline__ void async_ld16(void* lds, const void* gptr) {
  const unsigned l = (unsigned)(uintptr_t)lds;   // low 32 bits = LDS address
  asm volatile("global_load_async_to_lds_b128 %0, %1, off"
               :: "v"(l), "v"(gptr) : "memory");
}
__device__ __forceinline__ void async_wait0() {
  asm volatile("s_wait_asynccnt 0" ::: "memory");
}

// ---------------------------------------------------------------------------
// CDNA5 Tensor Data Mover: one descriptor DMAs a whole padded 2D tile to LDS.
// dim0 = contiguous elements (f16), dim1 = rows; LDS gets row padding so the
// padded stride matches the WMMA fragment layout strides.
// ---------------------------------------------------------------------------
__device__ __forceinline__ void tensor_wait0() {
#if __has_builtin(__builtin_amdgcn_s_wait_tensorcnt)
  __builtin_amdgcn_s_wait_tensorcnt((short)0);
#else
  asm volatile("s_wait_tensorcnt 0x0" ::: "memory");
#endif
}

#if HAVE_TDM
__device__ __forceinline__ void tdm_load_2d(void* lds, const void* gptr,
                                            unsigned dim0_rem, unsigned dim1_rem,
                                            unsigned tile0, unsigned tile1,
                                            unsigned stride0_elems,
                                            unsigned pad_int_code, unsigned pad_amt_code) {
  const unsigned long long ga = (unsigned long long)(uintptr_t)gptr;
  u32x4 g0;
  g0.x = 1u;                                   // count=1 valid descriptor
  g0.y = (unsigned)(uintptr_t)lds;             // LDS byte address
  g0.z = (unsigned)ga;                         // global addr [31:0]
  g0.w = (unsigned)((ga >> 32) & 0x1FFFFFFu)   // global addr [56:32]
         | 0x80000000u;                        // type=2 ("image")
  i32x8 g1;
  g1[0] = (int)((1u << 16)                     // data_size = 2 bytes
                | (1u << 20)                   // pad_enable
                | (pad_int_code << 22)         // pad every (2<<code) DWORDs
                | (pad_amt_code << 25));       // pad (code+1) DWORDs
  g1[1] = (int)((dim0_rem & 0xFFFFu) << 16);                               // tensor_dim0 lo
  g1[2] = (int)(((dim0_rem >> 16) & 0xFFFFu) | ((dim1_rem & 0xFFFFu) << 16));
  g1[3] = (int)(((dim1_rem >> 16) & 0xFFFFu) | (tile0 << 16));             // tile_dim0
  g1[4] = (int)tile1;                                                      // tile_dim1 (tile_dim2=0)
  g1[5] = (int)stride0_elems;                                              // tensor_dim0_stride lo
  g1[6] = 0;
  g1[7] = 0;
  i32x4 z4 = {0, 0, 0, 0};
#if defined(__clang_major__) && (__clang_major__ >= 23)
  i32x8 z8 = {0, 0, 0, 0, 0, 0, 0, 0};
  __builtin_amdgcn_tensor_load_to_lds(g0, g1, z4, z4, z8, 0);
#else
  __builtin_amdgcn_tensor_load_to_lds(g0, g1, z4, z4, 0);
#endif
}
#endif

// ---------------------------------------------------------------------------
// helpers
// ---------------------------------------------------------------------------
__device__ __forceinline__ v16h ld_frag(const _Float16* p0, const _Float16* p1) {
  union { v16h v; uint4 u[2]; } r;
  r.u[0] = *(const uint4*)p0;
  r.u[1] = *(const uint4*)p1;
  return r.v;
}

__device__ __forceinline__ float red_max16(float v) {
  v = fmaxf(v, __shfl_xor(v, 1));
  v = fmaxf(v, __shfl_xor(v, 2));
  v = fmaxf(v, __shfl_xor(v, 4));
  v = fmaxf(v, __shfl_xor(v, 8));
  return v;
}
__device__ __forceinline__ float red_sum16(float v) {
  v += __shfl_xor(v, 1);
  v += __shfl_xor(v, 2);
  v += __shfl_xor(v, 4);
  v += __shfl_xor(v, 8);
  return v;
}

// ---------------------------------------------------------------------------
// fp32 -> fp16 convert
// ---------------------------------------------------------------------------
__global__ void cvt_f32_f16(const float* __restrict__ src,
                            _Float16* __restrict__ dst, int n) {
  int i = blockIdx.x * blockDim.x + threadIdx.x;
  int stride = gridDim.x * blockDim.x;
  for (; i < n; i += stride) dst[i] = (_Float16)src[i];
}

// ---------------------------------------------------------------------------
// GEMM tile staging: one 128x32 K-slab of A and of W into padded LDS
// (stride 48 halves = 64B row + 32B pad).  TDM: 2 descriptors from wave 0.
// ---------------------------------------------------------------------------
__device__ __forceinline__ void gemm_stage(const _Float16* __restrict__ A,
                                           const _Float16* __restrict__ W,
                                           _Float16* AhB, _Float16* BhB,
                                           int m0, int n0, int M, int N, int K,
                                           int kk, int tid) {
#if HAVE_TDM
  if (tid < 32) {   // wave 0 issues the DMAs (scalar op, EXEC ignored)
    tdm_load_2d(AhB, &A[(size_t)m0 * K + kk], (unsigned)(K - kk),
                (unsigned)(M - m0), 32u, 128u, (unsigned)K, 3u, 7u);
    tdm_load_2d(BhB, &W[(size_t)n0 * K + kk], (unsigned)(K - kk),
                (unsigned)(N - n0), 32u, 128u, (unsigned)K, 3u, 7u);
  }
#else
#pragma unroll
  for (int it = 0; it < 2; ++it) {
    const int i   = tid + it * 256;
    const int row = i >> 2;
    const int cc  = (i & 3) * 8;
    async_ld16(&AhB[row * 48 + cc], &A[(size_t)(m0 + row) * K + kk + cc]);
    async_ld16(&BhB[row * 48 + cc], &W[(size_t)(n0 + row) * K + kk + cc]);
  }
#endif
}

__device__ __forceinline__ void stage_wait(int tid) {
#if HAVE_TDM
  if (tid < 32) tensor_wait0();
#else
  async_wait0();
#endif
}

// ---------------------------------------------------------------------------
// C[M,N] (f32) = A[M,K] (f16, row major) * W[N,K]^T (f16, row major)
// 128x128 tile, 8 waves, wave = 32x64, WMMA 16x16x32, TDM double buffer
// ---------------------------------------------------------------------------
__global__ __launch_bounds__(256)
void gemm_nt_f16(const _Float16* __restrict__ A,
                 const _Float16* __restrict__ W,
                 float* __restrict__ C, int M, int N, int K) {
  __shared__ _Float16 Ah[2][128 * 48];
  __shared__ _Float16 Bh[2][128 * 48];

  const int tid  = threadIdx.x;
  const int wave = tid >> 5;
  const int lane = tid & 31;
  const int ln16 = lane & 15;
  const int half = lane >> 4;

  const int m0 = blockIdx.y * 128;
  const int n0 = blockIdx.x * 128;
  const int wm = (wave & 3) * 32;
  const int wn = (wave >> 2) * 64;

  v8f acc[2][4];
#pragma unroll
  for (int mi = 0; mi < 2; ++mi)
#pragma unroll
    for (int ni = 0; ni < 4; ++ni) acc[mi][ni] = {};

  gemm_stage(A, W, Ah[0], Bh[0], m0, n0, M, N, K, 0, tid);

  for (int kk = 0; kk < K; kk += 32) {
    const int cur = (kk >> 5) & 1;
    stage_wait(tid);          // issuing wave's DMA of buffer `cur` complete
    __syncthreads();          // tile published to all waves
    if (kk + 32 < K)
      gemm_stage(A, W, Ah[cur ^ 1], Bh[cur ^ 1], m0, n0, M, N, K, kk + 32, tid);

    v16h af[2], bf[4];
#pragma unroll
    for (int mi = 0; mi < 2; ++mi) {
      const _Float16* p = &Ah[cur][(wm + mi * 16 + ln16) * 48];
      af[mi] = ld_frag(p + half * 8, p + 16 + half * 8);
    }
#pragma unroll
    for (int ni = 0; ni < 4; ++ni) {
      const _Float16* p = &Bh[cur][(wn + ni * 16 + ln16) * 48];
      bf[ni] = ld_frag(p + half * 8, p + 16 + half * 8);
    }
#pragma unroll
    for (int mi = 0; mi < 2; ++mi)
#pragma unroll
      for (int ni = 0; ni < 4; ++ni)
        acc[mi][ni] = __builtin_amdgcn_wmma_f32_16x16x32_f16(
            false, af[mi], false, bf[ni], (short)0, acc[mi][ni], false, false);

    __syncthreads();          // all reads of `cur` done before refill
  }

#pragma unroll
  for (int mi = 0; mi < 2; ++mi)
#pragma unroll
    for (int ni = 0; ni < 4; ++ni) {
#pragma unroll
      for (int r = 0; r < 8; ++r) {
        const int row = m0 + wm + mi * 16 + r + 8 * half;
        const int col = n0 + wn + ni * 16 + ln16;
        C[(size_t)row * N + col] = acc[mi][ni][r];
      }
    }
}

// ---------------------------------------------------------------------------
// RoPE (optional) + scale + transpose [b][s][h][dk] -> [b][h][s][dk], f16 out
// ---------------------------------------------------------------------------
__global__ void rope_cvt(const float* __restrict__ src,
                         const int* __restrict__ pos,
                         _Float16* __restrict__ dst,
                         float scale, int do_rope) {
  const int idx = blockIdx.x * blockDim.x + threadIdx.x;
  const int i2 = idx & 63;
  const int h  = (idx >> 6) & 15;
  const int s  = (idx >> 10) & 2047;
  const int b  = idx >> 21;
  if (b >= Bx) return;

  const size_t sbase = ((size_t)(b * Sx + s)) * Dx + h * DKx + 2 * i2;
  float x1 = src[sbase];
  float x2 = src[sbase + 1];
  float r1 = x1, r2 = x2;
  if (do_rope) {
    const float p    = (float)pos[s];
    const float freq = __powf(10000.0f, -(2.0f * (float)i2) / 128.0f);
    float sn, cs;
    __sincosf(p * freq, &sn, &cs);
    r1 = x1 * cs - x2 * sn;
    r2 = x1 * sn + x2 * cs;
  }
  const size_t dbase = (((size_t)(b * Hx + h)) * Sx + s) * DKx + 2 * i2;
  dst[dbase]     = (_Float16)(r1 * scale);
  dst[dbase + 1] = (_Float16)(r2 * scale);
}

// ---------------------------------------------------------------------------
// Attention tile staging: 32x128 K tile and V tile into padded LDS
// (stride 136 halves = 256B row + 16B pad).  TDM: 2 descriptors from wave 0.
// ---------------------------------------------------------------------------
__device__ __forceinline__ void attn_stage(const _Float16* __restrict__ Kh,
                                           const _Float16* __restrict__ Vh,
                                           _Float16* KtB, _Float16* VsB,
                                           size_t bhS, int k0, int tid) {
#if HAVE_TDM
  if (tid < 32) {
    tdm_load_2d(KtB, &Kh[(bhS + k0) << 7], 128u, (unsigned)(Sx - k0),
                128u, 32u, 128u, 5u, 3u);
    tdm_load_2d(VsB, &Vh[(bhS + k0) << 7], 128u, (unsigned)(Sx - k0),
                128u, 32u, 128u, 5u, 3u);
  }
#else
#pragma unroll
  for (int it = 0; it < 4; ++it) {
    const int i   = tid + it * 128;
    const int row = i >> 4;
    const int cc  = (i & 15) * 8;
    const size_t g = ((bhS + k0 + row) << 7) + cc;
    async_ld16(&KtB[row * 136 + cc], &Kh[g]);
    async_ld16(&VsB[row * 136 + cc], &Vh[g]);
  }
#endif
}

// ---------------------------------------------------------------------------
// Flash attention: grid (S/64, B*H), 128 threads (4 waves), 16 q-rows/wave.
// TDM double-buffered K/V staging; V transposed LDS->LDS for PV B-frags.
// ---------------------------------------------------------------------------
__global__ __launch_bounds__(128)
void flash_attn(const _Float16* __restrict__ Qh,
                const _Float16* __restrict__ Kh,
                const _Float16* __restrict__ Vh,
                _Float16* __restrict__ Oh) {
  __shared__ _Float16 Kt[2][32 * 136];
  __shared__ _Float16 Vs[2][32 * 136];
  __shared__ _Float16 Vt[128 * 48];
  __shared__ _Float16 Pl[4 * 16 * 48];

  const int tid  = threadIdx.x;
  const int wave = tid >> 5;
  const int lane = tid & 31;
  const int ln16 = lane & 15;
  const int half = lane >> 4;

  const int qblk = blockIdx.x;
  const int bh   = blockIdx.y;
  const size_t bhS = (size_t)bh * Sx;
  const int qtile = qblk * 64 + wave * 16;

  v16h qf[4];
  {
    const _Float16* qrow = Qh + ((bhS + qtile + ln16) << 7);
#pragma unroll
    for (int c = 0; c < 4; ++c)
      qf[c] = ld_frag(qrow + c * 32 + half * 8, qrow + c * 32 + 16 + half * 8);
  }

  float mrow[8], lrow[8];
  v8f acc[8];
#pragma unroll
  for (int r = 0; r < 8; ++r) { mrow[r] = -3.0e38f; lrow[r] = 0.0f; }
#pragma unroll
  for (int dt = 0; dt < 8; ++dt) acc[dt] = {};

  const int nkb = (qblk + 1) * 2;
  attn_stage(Kh, Vh, Kt[0], Vs[0], bhS, 0, tid);

  for (int kb = 0; kb < nkb; ++kb) {
    const int cur = kb & 1;
    stage_wait(tid);
    __syncthreads();
    if (kb + 1 < nkb)
      attn_stage(Kh, Vh, Kt[cur ^ 1], Vs[cur ^ 1], bhS, (kb + 1) << 5, tid);

    // transpose Vs[cur] (32x128) -> Vt (128 x 48-stride)
#pragma unroll
    for (int it = 0; it < 4; ++it) {
      const int i   = tid + it * 128;
      const int row = i >> 4;
      const int cc  = (i & 15) * 8;
      union { uint4 u; _Float16 h8[8]; } t;
      t.u = *(const uint4*)&Vs[cur][row * 136 + cc];
#pragma unroll
      for (int j = 0; j < 8; ++j) Vt[(cc + j) * 48 + row] = t.h8[j];
    }
    __syncthreads();

    const int k0 = kb << 5;

    // ---- scores ----
    v8f sfr[2];
#pragma unroll
    for (int kt = 0; kt < 2; ++kt) {
      v8f s = {};
      const _Float16* krow = &Kt[cur][(kt * 16 + ln16) * 136];
#pragma unroll
      for (int c = 0; c < 4; ++c) {
        v16h b = ld_frag(krow + c * 32 + half * 8, krow + c * 32 + 16 + half * 8);
        s = __builtin_amdgcn_wmma_f32_16x16x32_f16(false, qf[c], false, b,
                                                   (short)0, s, false, false);
      }
      const int keyabs = k0 + kt * 16 + ln16;
#pragma unroll
      for (int r = 0; r < 8; ++r) {
        const int qabs = qtile + r + 8 * half;
        s[r] = (keyabs > qabs) ? -3.0e38f : s[r];
      }
      sfr[kt] = s;
    }

    // ---- online softmax ----
    float corr[8];
#pragma unroll
    for (int r = 0; r < 8; ++r) {
      float mx = red_max16(fmaxf(sfr[0][r], sfr[1][r]));
      const float mnew = fmaxf(mrow[r], mx);
      const float c  = exp2f((mrow[r] - mnew) * 1.4426950408889634f);
      const float p0 = exp2f((sfr[0][r] - mnew) * 1.4426950408889634f);
      const float p1 = exp2f((sfr[1][r] - mnew) * 1.4426950408889634f);
      sfr[0][r] = p0;
      sfr[1][r] = p1;
      lrow[r] = lrow[r] * c + red_sum16(p0 + p1);
      mrow[r] = mnew;
      corr[r] = c;
    }
#pragma unroll
    for (int dt = 0; dt < 8; ++dt)
#pragma unroll
      for (int r = 0; r < 8; ++r) acc[dt][r] *= corr[r];

    // ---- P: C-frag -> A-frag re-swizzle via per-wave LDS patch ----
    _Float16* Pw = &Pl[wave * 16 * 48];
#pragma unroll
    for (int kt = 0; kt < 2; ++kt)
#pragma unroll
      for (int r = 0; r < 8; ++r)
        Pw[(r + 8 * half) * 48 + kt * 16 + ln16] = (_Float16)sfr[kt][r];

    const _Float16* prow = &Pw[ln16 * 48];
    v16h pa = ld_frag(prow + half * 8, prow + 16 + half * 8);

    // ---- PV ----
#pragma unroll
    for (int dt = 0; dt < 8; ++dt) {
      const _Float16* vrow = &Vt[(dt * 16 + ln16) * 48];
      v16h vf = ld_frag(vrow + half * 8, vrow + 16 + half * 8);
      acc[dt] = __builtin_amdgcn_wmma_f32_16x16x32_f16(false, pa, false, vf,
                                                       (short)0, acc[dt], false, false);
    }
    __syncthreads();
  }

  const int b = bh >> 4;
  const int h = bh & 15;
#pragma unroll
  for (int dt = 0; dt < 8; ++dt)
#pragma unroll
    for (int r = 0; r < 8; ++r) {
      const int q = qtile + r + 8 * half;
      const float o = acc[dt][r] / lrow[r];
      Oh[((size_t)b * Sx + q) * Dx + h * DKx + dt * 16 + ln16] = (_Float16)o;
    }
}

// ---------------------------------------------------------------------------
// host-side orchestration
// ---------------------------------------------------------------------------
extern "C" void kernel_launch(void* const* d_in, const int* in_sizes, int n_in,
                              void* d_out, int out_size, void* d_ws, size_t ws_size,
                              hipStream_t stream) {
  (void)in_sizes; (void)n_in; (void)out_size; (void)ws_size;

  const float* x  = (const float*)d_in[0];
  const int*   tp = (const int*)d_in[1];
  const float* Wq = (const float*)d_in[2];
  const float* Wk = (const float*)d_in[3];
  const float* Wv = (const float*)d_in[4];
  const float* Wo = (const float*)d_in[5];
  float* out = (float*)d_out;
  char*  ws  = (char*)d_ws;

  const size_t NX = (size_t)Mtot * Dx;
  const size_t NW = (size_t)Dx * Dx;

  _Float16* xh   = (_Float16*)(ws);
  _Float16* Wqh  = (_Float16*)(ws + 16777216);
  _Float16* Wkh  = (_Float16*)(ws + 25165824);
  _Float16* Wvh  = (_Float16*)(ws + 33554432);
  _Float16* Woh  = (_Float16*)(ws + 41943040);
  float*    Qf   = (float*)   (ws + 50331648);
  float*    Kf   = (float*)   (ws + 83886080);
  float*    Vf   = (float*)   (ws + 117440512);
  _Float16* Qh2  = (_Float16*)(ws + 150994944);
  _Float16* Kh2  = (_Float16*)(ws + 167772160);
  _Float16* Vh2  = (_Float16*)(ws + 184549376);
  _Float16* AttH = (_Float16*)(ws + 201326592);

  cvt_f32_f16<<<4096, 256, 0, stream>>>(x,  xh,  (int)NX);
  cvt_f32_f16<<<2048, 256, 0, stream>>>(Wq, Wqh, (int)NW);
  cvt_f32_f16<<<2048, 256, 0, stream>>>(Wk, Wkh, (int)NW);
  cvt_f32_f16<<<2048, 256, 0, stream>>>(Wv, Wvh, (int)NW);
  cvt_f32_f16<<<2048, 256, 0, stream>>>(Wo, Woh, (int)NW);

  dim3 ggrid(Dx / 128, Mtot / 128);
  gemm_nt_f16<<<ggrid, 256, 0, stream>>>(xh, Wqh, Qf, Mtot, Dx, Dx);
  gemm_nt_f16<<<ggrid, 256, 0, stream>>>(xh, Wkh, Kf, Mtot, Dx, Dx);
  gemm_nt_f16<<<ggrid, 256, 0, stream>>>(xh, Wvh, Vf, Mtot, Dx, Dx);

  const int ropeN = Bx * Sx * Hx * 64;
  rope_cvt<<<ropeN / 256, 256, 0, stream>>>(Qf, tp, Qh2, 0.08838834764831845f, 1);
  rope_cvt<<<ropeN / 256, 256, 0, stream>>>(Kf, tp, Kh2, 1.0f, 1);
  rope_cvt<<<ropeN / 256, 256, 0, stream>>>(Vf, tp, Vh2, 1.0f, 0);

  flash_attn<<<dim3(Sx / 64, Bx * Hx), 128, 0, stream>>>(Qh2, Kh2, Vh2, AttH);

  gemm_nt_f16<<<ggrid, 256, 0, stream>>>(AttH, Woh, out, Mtot, Dx, Dx);
}
